// LSTM_20401094656583
// MI455X (gfx1250) — matrix operand order repaired and checked by reading, not compiled, not run on hardware
//
#include <hip/hip_runtime.h>

#define BATCH  4096
#define HDIM   64
#define EDIM   28
#define EMDIM  14
#define GATES  256   // 4*HDIM
#define POOLB  128
#define POOLH1 512
#define OBSLEN 16
#define PREDL  14
#define NGROUP 256
#define GSZ    16

typedef __attribute__((ext_vector_type(16))) __bf16          v16bf;
typedef __attribute__((ext_vector_type(16))) unsigned short  v16us;
typedef __attribute__((ext_vector_type(8)))  float           v8f;
typedef __attribute__((ext_vector_type(4)))  unsigned int    v4u;

static __device__ __forceinline__ unsigned short f2bf(float f) {
  unsigned u = __builtin_bit_cast(unsigned, f);
  u += 0x7FFFu + ((u >> 16) & 1u);           // round-to-nearest-even
  return (unsigned short)(u >> 16);
}
static __device__ __forceinline__ unsigned pack2bf(float a, float b) {
  return (unsigned)f2bf(a) | ((unsigned)f2bf(b) << 16);
}
static __device__ __forceinline__ float sigmoidf_(float x) {
  return 1.0f / (1.0f + __expf(-x));
}
static __device__ __forceinline__ unsigned lds_off(const void* p) {
  return (unsigned)(unsigned long long)p;    // LDS aperture: low 32 bits = LDS offset
}
// CDNA5 async global->LDS copy (ASYNCcnt-tracked, bypasses VGPRs)
static __device__ __forceinline__ void async_copy_b128(unsigned loff, const void* gaddr) {
  asm volatile("global_load_async_to_lds_b128 %0, %1, off"
               :: "v"(loff), "v"(gaddr) : "memory");
}
static __device__ __forceinline__ void wait_async0() {
  asm volatile("s_wait_asynccnt 0x0" ::: "memory");
}

// A fragment: base points at row start + k0 inside a [row][>=32] bf16 tile.
// lane needs k-runs {khalf*8..+8} and {16+khalf*8..+8}  ->  2x ds_load_b128
static __device__ __forceinline__ v16bf frag_a(const unsigned short* base, int khalf) {
  union { v16us v; v4u q[2]; } u;
  u.q[0] = *(const v4u*)(base + khalf * 8);
  u.q[1] = *(const v4u*)(base + 16 + khalf * 8);
  return __builtin_bit_cast(v16bf, u.v);
}
// B fragment: base points at n-row start + k0 inside a transposed [n][K] bf16 tile.
// lane needs k-runs {khalf*8..+8} and {16+khalf*8..+8}  ->  2x ds_load_b128
static __device__ __forceinline__ v16bf frag_b(const unsigned short* base, int khalf) {
  union { v16us v; v4u q[2]; } u;
  const unsigned short* pp = base + khalf * 8;
  u.q[0] = *(const v4u*)(pp);
  u.q[1] = *(const v4u*)(pp + 16);
  return __builtin_bit_cast(v16bf, u.v);
}

// ---------------------------------------------------------------------------
// Generic GEMM: C[M,N](f32) = act( A[M,K](f32->bf16) @ BT(bf16,[N][Kpad]) + bias )
// flags bit0: accumulate into existing C, bit1: relu
// 128 threads = 4 waves; 64x64 tile; K streamed in 32-chunks through LDS.
// B tile staged with async global->LDS copies; A tile converted via VALU.
// ---------------------------------------------------------------------------
__global__ void __launch_bounds__(128)
gemm_bf16_kernel(const float* __restrict__ A, int lda,
                 const unsigned short* __restrict__ BT,   // [N][Kpad] bf16
                 const float* __restrict__ bias,
                 float* __restrict__ C, int ldc,
                 int M, int N, int K, int flags)
{
  __shared__ unsigned short sA[64 * 32];    // [row][k]  row-major
  __shared__ unsigned short sB[64 * 32];    // [n][k]    k-contiguous
  const int tid   = threadIdx.x;
  const int wave  = tid >> 5;
  const int lane  = tid & 31;
  const int mlane = lane & 15;
  const int khalf = lane >> 4;
  const int bm = blockIdx.y * 64;
  const int bn = blockIdx.x * 64;
  const int Kpad = (K + 31) & ~31;

  const v8f zero8 = {0.f,0.f,0.f,0.f,0.f,0.f,0.f,0.f};
  v8f acc[4];
  acc[0] = zero8; acc[1] = zero8; acc[2] = zero8; acc[3] = zero8;

  const int kTiles = Kpad >> 5;
  for (int kt = 0; kt < kTiles; ++kt) {
    const int k0 = kt << 5;
    // stage B tile transposed: async 16B global->LDS copies from packed [N][Kpad]
    for (int i = tid; i < 64 * 4; i += 128) {
      int c = i >> 2, seg = i & 3;
      int gc = bn + c;
      unsigned short* dst = sB + c * 32 + seg * 8;
      if (gc < N) {
        async_copy_b128(lds_off(dst), BT + (size_t)gc * Kpad + k0 + seg * 8);
      } else {
        v4u zz = {0u,0u,0u,0u};
        *(v4u*)dst = zz;
      }
    }
    // stage A tile: pack 2 bf16 per b32 LDS store (needs VALU convert)
    for (int i = tid; i < 64 * 16; i += 128) {
      int r = i >> 4, cp = (i & 15) << 1;
      int gr = bm + r, gc = k0 + cp;
      float v0 = 0.0f, v1 = 0.0f;
      if (gr < M) {
        if (gc     < K) v0 = A[(size_t)gr * lda + gc];
        if (gc + 1 < K) v1 = A[(size_t)gr * lda + gc + 1];
      }
      *(unsigned*)(sA + r * 32 + cp) = pack2bf(v0, v1);
    }
    if (kt + 1 < kTiles) {                   // warm next A k-tile (global_prefetch_b8)
      int gr = bm + (tid >> 1);
      if (gr < M && k0 + 32 < K) __builtin_prefetch(&A[(size_t)gr * lda + k0 + 32], 0, 1);
    }
    wait_async0();
    __syncthreads();

    const int rowA = (wave << 4) + mlane;
    v16bf afrag = frag_a(sA + rowA * 32, khalf);
#pragma unroll
    for (int t = 0; t < 4; ++t) {
      v16bf bfrag = frag_b(sB + ((t << 4) + mlane) * 32, khalf);
      acc[t] = __builtin_amdgcn_wmma_f32_16x16x32_bf16(
          false, afrag, false, bfrag, (short)0, acc[t], false, false);
    }
    __syncthreads();
  }

  // C/D layout: VGPR r -> row r (lanes 0-15) / row 8+r (lanes 16-31), col = lane&15
  const int row0 = bm + (wave << 4) + (khalf << 3);
#pragma unroll
  for (int t = 0; t < 4; ++t) {
    int col = bn + (t << 4) + mlane;
    if (col >= N) continue;
    float bv = bias ? bias[col] : 0.0f;
#pragma unroll
    for (int r = 0; r < 8; ++r) {
      int row = row0 + r;
      if (row >= M) continue;
      float v = acc[t][r] + bv;
      if (flags & 1) v += C[(size_t)row * ldc + col];
      if (flags & 2) v = fmaxf(v, 0.0f);
      C[(size_t)row * ldc + col] = v;
    }
  }
}

// ---------------------------------------------------------------------------
// Fused social-pool kernel. One block = one group of 16 agents.
// X[256x96] = [pos_j - pos_i | h_j | 0pad]  built in LDS as bf16
// L1: relu(X @ W1^T + b1) -> [256x512], streamed in 64-col chunks
// L2: relu(Y @ W2^T + b2) -> [256x128], accumulated in registers
// pool[i] = max_j Z[i*16+j]  (wave w owns exactly the 16 rows of agent i=w)
// 512 threads = 16 waves. ~96KB dynamic LDS (CDNA5: 320KB/WGP).
// Weight chunks streamed with async global->LDS copies.
// ---------------------------------------------------------------------------
__global__ void __launch_bounds__(512)
pool_kernel(const float* __restrict__ pose,           // [4096 x 28]
            const float* __restrict__ h,              // [4096 x 64]
            const unsigned short* __restrict__ PW1t,  // [512][96] bf16
            const float* __restrict__ b1,             // [512]
            const unsigned short* __restrict__ PW2t,  // [128][512] bf16
            const float* __restrict__ b2,             // [128]
            float* __restrict__ pool_out)             // [4096 x 128]
{
  extern __shared__ unsigned short smem[];
  unsigned short* sX = smem;                 // [256][96] row-major (A for L1)
  unsigned short* sY = smem + 256 * 96;      // [256][64] row-major (A for L2)
  unsigned short* sW = sY + 256 * 64;        // transposed B chunk, up to [128][64]
  const int tid   = threadIdx.x;
  const int wave  = tid >> 5;
  const int lane  = tid & 31;
  const int mlane = lane & 15;
  const int khalf = lane >> 4;
  const int gbase = blockIdx.x * GSZ;

  // build X, two bf16 per b32 store
  for (int idx = tid; idx < 256 * 48; idx += 512) {
    int r = idx / 48, cp = (idx % 48) * 2;
    int i = r >> 4, j = r & 15;
    float v0 = 0.0f, v1 = 0.0f;
    if (cp < EDIM) {
      v0 = pose[(size_t)(gbase + j) * EDIM + cp]     - pose[(size_t)(gbase + i) * EDIM + cp];
      v1 = pose[(size_t)(gbase + j) * EDIM + cp + 1] - pose[(size_t)(gbase + i) * EDIM + cp + 1];
    } else if (cp < 92) {
      v0 = h[(size_t)(gbase + j) * HDIM + (cp - EDIM)];
      v1 = (cp + 1 < 92) ? h[(size_t)(gbase + j) * HDIM + (cp + 1 - EDIM)] : 0.0f;
    }
    *(unsigned*)(sX + r * 96 + cp) = pack2bf(v0, v1);
  }
  __syncthreads();

  const v8f zero8 = {0.f,0.f,0.f,0.f,0.f,0.f,0.f,0.f};
  v8f accZ[8];
#pragma unroll
  for (int t = 0; t < 8; ++t) accZ[t] = zero8;

  const int rowA = (wave << 4) + mlane;

  for (int ch = 0; ch < 8; ++ch) {           // layer-1 output chunk of 64 cols
    const int n0 = ch * 64;
    // stage W1 chunk transposed [64][96]: async 16B copies
    for (int i = tid; i < 64 * 12; i += 512) {
      int n = i / 12, seg = i % 12;
      async_copy_b128(lds_off(sW + n * 96 + seg * 8),
                      PW1t + (size_t)(n0 + n) * 96 + seg * 8);
    }
    wait_async0();
    __syncthreads();

    v8f accY[4];
#pragma unroll
    for (int t = 0; t < 4; ++t) accY[t] = zero8;
#pragma unroll
    for (int kt = 0; kt < 3; ++kt) {         // K = 96
      const int k0 = kt << 5;
      v16bf afrag = frag_a(sX + rowA * 96 + k0, khalf);
#pragma unroll
      for (int t = 0; t < 4; ++t) {
        v16bf bfrag = frag_b(sW + ((t << 4) + mlane) * 96 + k0, khalf);
        accY[t] = __builtin_amdgcn_wmma_f32_16x16x32_bf16(
            false, afrag, false, bfrag, (short)0, accY[t], false, false);
      }
    }
    __syncthreads();                          // done reading sW (W1 chunk)

    // relu(Y + b1) -> sY as bf16 (each wave writes only its own 16 rows)
#pragma unroll
    for (int t = 0; t < 4; ++t) {
      int colY = (t << 4) + mlane;
      float bb = b1[n0 + colY];
#pragma unroll
      for (int r = 0; r < 8; ++r) {
        int rowY = (wave << 4) + (khalf << 3) + r;
        sY[rowY * 64 + colY] = f2bf(fmaxf(accY[t][r] + bb, 0.0f));
      }
    }
    // stage W2 chunk transposed [128][64] for K-slice [n0, n0+64): async copies
    for (int i = tid; i < 128 * 8; i += 512) {
      int n = i >> 3, seg = i & 7;
      async_copy_b128(lds_off(sW + n * 64 + seg * 8),
                      PW2t + (size_t)n * POOLH1 + n0 + seg * 8);
    }
    wait_async0();
    __syncthreads();

#pragma unroll
    for (int kt = 0; kt < 2; ++kt) {          // layer-2 partial, K-chunk = 64
      const int k0 = kt << 5;
      v16bf afrag = frag_a(sY + rowA * 64 + k0, khalf);
#pragma unroll
      for (int t = 0; t < 8; ++t) {
        v16bf bfrag = frag_b(sW + ((t << 4) + mlane) * 64 + k0, khalf);
        accZ[t] = __builtin_amdgcn_wmma_f32_16x16x32_bf16(
            false, afrag, false, bfrag, (short)0, accZ[t], false, false);
      }
    }
    __syncthreads();                          // before next chunk reuses sW/sY
  }

  // epilogue: relu(Z+b2), max over the 16 j-rows of this wave's tile (i = wave)
#pragma unroll
  for (int t = 0; t < 8; ++t) {
    int col  = (t << 4) + mlane;
    float bb = b2[col];
    float mx = 0.0f;                          // relu outputs are >= 0
#pragma unroll
    for (int r = 0; r < 8; ++r)
      mx = fmaxf(mx, fmaxf(accZ[t][r] + bb, 0.0f));
    float other = __shfl_xor(mx, 16, 32);     // combine rows 0-7 with 8-15
    mx = fmaxf(mx, other);
    if (khalf == 0)
      pool_out[(size_t)(gbase + wave) * POOLB + col] = mx;
  }
}

// ---------------------------------------------------------------------------
// Pointwise kernels
// ---------------------------------------------------------------------------
__global__ void pack_bt_kernel(const float* __restrict__ W,      // [N x ldk] f32
                               unsigned short* __restrict__ BT,  // [N][Kpad] bf16
                               int N, int K, int Kpad, int ldk, int koff) {
  int idx = blockIdx.x * blockDim.x + threadIdx.x;
  if (idx >= N * Kpad) return;
  int n = idx / Kpad, k = idx % Kpad;
  BT[idx] = (k < K) ? f2bf(W[(size_t)n * ldk + koff + k]) : (unsigned short)0;
}

__global__ void add_vec_kernel(const float* a, const float* b, float* o, int n) {
  int i = blockIdx.x * blockDim.x + threadIdx.x;
  if (i < n) o[i] = a[i] + b[i];
}

__global__ void zero_f32_kernel(float* p, int n) {
  int i = blockIdx.x * blockDim.x + threadIdx.x;
  if (i < n) p[i] = 0.0f;
}

__global__ void lstm_update_kernel(const float* __restrict__ gates,  // [B x 256] i,f,g,o
                                   float* __restrict__ h, float* __restrict__ c, int n) {
  int idx = blockIdx.x * blockDim.x + threadIdx.x;
  if (idx >= n) return;
  int b = idx / HDIM, k = idx % HDIM;
  const float* z = gates + (size_t)b * GATES;
  float si = sigmoidf_(z[k]);
  float sf = sigmoidf_(z[HDIM + k]);
  float tg = tanhf(z[2 * HDIM + k]);
  float so = sigmoidf_(z[3 * HDIM + k]);
  float cc = sf * c[idx] + si * tg;
  c[idx] = cc;
  h[idx] = so * tanhf(cc);
}

__global__ void bn_relu_kernel(const float* __restrict__ x, float* __restrict__ y,
                               const float* __restrict__ g, const float* __restrict__ be,
                               const float* __restrict__ m, const float* __restrict__ v,
                               int n, int D) {
  int idx = blockIdx.x * blockDim.x + threadIdx.x;
  if (idx >= n) return;
  int d = idx % D;
  float t = (x[idx] - m[d]) * rsqrtf(v[d] + 1e-5f) * g[d] + be[d];
  y[idx] = fmaxf(t, 0.0f);
}

__global__ void h2p_epi_kernel(const float* __restrict__ sraw, float* __restrict__ last_speed,
                               float* __restrict__ pose, float* __restrict__ out_speed, int n) {
  int idx = blockIdx.x * blockDim.x + threadIdx.x;
  if (idx >= n) return;
  float s = fminf(fmaxf(sraw[idx], -190.0f), 190.0f);
  last_speed[idx] = s;
  out_speed[idx]  = s;
  pose[idx]      += s;
}

__global__ void h2m_epi_kernel(const float* __restrict__ mraw, const float* __restrict__ pose,
                               float* __restrict__ last_mask, float* __restrict__ out_mask, int n) {
  int idx = blockIdx.x * blockDim.x + threadIdx.x;
  if (idx >= n) return;
  float sg = sigmoidf_(mraw[idx]);
  last_mask[idx] = sg;                       // raw sigmoid feeds next dec_m step
  int b = idx / EMDIM, c = idx % EMDIM;
  float dx = pose[(size_t)b * EDIM + 2 * c];
  float dy = pose[(size_t)b * EDIM + 2 * c + 1];
  float d  = sqrtf(dx * dx + dy * dy);
  out_mask[idx] = (d < 50.0f) ? 0.01f : sg;  // thresholded value goes to output
}

// ---------------------------------------------------------------------------
// Host orchestration
// ---------------------------------------------------------------------------
struct Ptrs {
  const float *obs_pose, *obs_speed, *obs_mask;
  const float *ep_wih, *ep_whh, *ep_bih, *ep_bhh;
  const float *dp_wih, *dp_whh, *dp_bih, *dp_bhh;
  const float *em_wih, *em_whh, *em_bih, *em_bhh;
  const float *dm_wih, *dm_whh, *dm_bih, *dm_bhh;
  const float *h2p_w, *h2p_b, *h2m_w, *h2m_b, *l1_w, *l1_b;
  const float *pool_w1, *pool_b1, *pool_w2, *pool_b2;
  const float *m_w1, *m_b1, *m_g1, *m_be1, *m_m1, *m_v1;
  const float *m_w2, *m_b2, *m_g2, *m_be2, *m_m2, *m_v2;
};

static inline int upk(int k) { return (k + 31) & ~31; }

static void launch_gemm(hipStream_t s, const float* A, int lda, const unsigned short* BT,
                        const float* bias, float* C, int ldc, int M, int N, int K, int flags) {
  dim3 grid((N + 63) / 64, (M + 63) / 64);
  gemm_bf16_kernel<<<grid, 128, 0, s>>>(A, lda, BT, bias, C, ldc, M, N, K, flags);
}
static void launch_pack(hipStream_t s, const float* W, unsigned short* BT,
                        int N, int K, int Kpad, int ldk, int koff) {
  int n = N * Kpad;
  pack_bt_kernel<<<(n + 255) / 256, 256, 0, s>>>(W, BT, N, K, Kpad, ldk, koff);
}

extern "C" void kernel_launch(void* const* d_in, const int* in_sizes, int n_in,
                              void* d_out, int out_size, void* d_ws, size_t ws_size,
                              hipStream_t stream) {
  const float* in[64];
  for (int i = 0; i < n_in && i < 64; ++i) in[i] = (const float*)d_in[i];

  Ptrs p;
  if (in_sizes[0] == OBSLEN * BATCH * EDIM) {
    // convention A: setup_inputs() insertion order, params traversed in insertion order
    p.obs_pose = in[0];  p.obs_speed = in[1]; p.obs_mask = in[2];
    p.ep_wih = in[3];  p.ep_whh = in[4];  p.ep_bih = in[5];  p.ep_bhh = in[6];
    p.dp_wih = in[7];  p.dp_whh = in[8];  p.dp_bih = in[9];  p.dp_bhh = in[10];
    p.em_wih = in[11]; p.em_whh = in[12]; p.em_bih = in[13]; p.em_bhh = in[14];
    p.dm_wih = in[15]; p.dm_whh = in[16]; p.dm_bih = in[17]; p.dm_bhh = in[18];
    p.h2p_w = in[19]; p.h2p_b = in[20]; p.h2m_w = in[21]; p.h2m_b = in[22];
    p.l1_w = in[23]; p.l1_b = in[24];
    p.pool_w1 = in[25]; p.pool_b1 = in[26]; p.pool_w2 = in[27]; p.pool_b2 = in[28];
    p.m_w1 = in[29]; p.m_b1 = in[30]; p.m_g1 = in[31]; p.m_be1 = in[32];
    p.m_m1 = in[33]; p.m_v1 = in[34]; p.m_w2 = in[35]; p.m_b2 = in[36];
    p.m_g2 = in[37]; p.m_be2 = in[38]; p.m_m2 = in[39]; p.m_v2 = in[40];
  } else {
    // convention B: JAX tree flatten (recursively sorted dict keys)
    p.obs_mask = in[0]; p.obs_pose = in[1]; p.obs_speed = in[2];
    p.dm_bhh = in[3];  p.dm_bih = in[4];  p.dm_whh = in[5];  p.dm_wih = in[6];
    p.dp_bhh = in[7];  p.dp_bih = in[8];  p.dp_whh = in[9];  p.dp_wih = in[10];
    p.em_bhh = in[11]; p.em_bih = in[12]; p.em_whh = in[13]; p.em_wih = in[14];
    p.ep_bhh = in[15]; p.ep_bih = in[16]; p.ep_whh = in[17]; p.ep_wih = in[18];
    p.h2m_b = in[19]; p.h2m_w = in[20]; p.h2p_b = in[21]; p.h2p_w = in[22];
    p.l1_b = in[23]; p.l1_w = in[24];
    p.m_b1 = in[25]; p.m_b2 = in[26]; p.m_be1 = in[27]; p.m_be2 = in[28];
    p.m_g1 = in[29]; p.m_g2 = in[30]; p.m_m1 = in[31]; p.m_m2 = in[32];
    p.m_v1 = in[33]; p.m_v2 = in[34]; p.m_w1 = in[35]; p.m_w2 = in[36];
    p.pool_b1 = in[37]; p.pool_b2 = in[38]; p.pool_w1 = in[39]; p.pool_w2 = in[40];
  }

  // ---- workspace bump allocator ----
  char* wsb = (char*)d_ws;
  size_t off = 0;
  auto allocf = [&](size_t n) -> float* {
    float* q = (float*)(wsb + off); off += ((n * 4) + 255) & ~(size_t)255; return q;
  };
  auto allocu = [&](size_t n) -> unsigned short* {
    unsigned short* q = (unsigned short*)(wsb + off); off += ((n * 2) + 255) & ~(size_t)255; return q;
  };

  float* h_p  = allocf((size_t)BATCH * HDIM);
  float* c_p  = allocf((size_t)BATCH * HDIM);
  float* hmA  = allocf((size_t)BATCH * HDIM);
  float* hmB  = allocf((size_t)BATCH * HDIM);
  float* c_m  = allocf((size_t)BATCH * HDIM);
  float* gates= allocf((size_t)BATCH * GATES);
  float* poolh= allocf((size_t)BATCH * POOLB);
  float* xbuf = allocf((size_t)BATCH * 128);
  float* tmp64= allocf((size_t)BATCH * HDIM);
  float* cpose= allocf((size_t)BATCH * EDIM);
  float* lspd = allocf((size_t)BATCH * EDIM);
  float* lmask= allocf((size_t)BATCH * EMDIM);
  float* sraw = allocf((size_t)BATCH * EDIM);
  float* mraw = allocf((size_t)BATCH * EMDIM);
  float* b_ep = allocf(GATES); float* b_em = allocf(GATES);
  float* b_dp = allocf(GATES); float* b_dm = allocf(GATES);

  // packed bf16 weights, layout [N][Kpad]
  unsigned short* wihP = allocu((size_t)GATES * upk(EDIM));
  unsigned short* whhP = allocu((size_t)GATES * upk(HDIM));
  unsigned short* wihM = allocu((size_t)GATES * upk(EMDIM));
  unsigned short* whhM = allocu((size_t)GATES * upk(HDIM));
  unsigned short* dwihP= allocu((size_t)GATES * upk(EDIM));
  unsigned short* dwhhP= allocu((size_t)GATES * upk(HDIM));
  unsigned short* dwihM= allocu((size_t)GATES * upk(EMDIM));
  unsigned short* dwhhM= allocu((size_t)GATES * upk(HDIM));
  unsigned short* H2P  = allocu((size_t)EDIM  * upk(HDIM));
  unsigned short* H2M  = allocu((size_t)EMDIM * upk(HDIM));
  unsigned short* L1a  = allocu((size_t)HDIM * upk(HDIM));
  unsigned short* L1b  = allocu((size_t)HDIM * upk(HDIM));
  unsigned short* W1a  = allocu((size_t)128 * upk(HDIM));
  unsigned short* W1b  = allocu((size_t)128 * upk(POOLB));
  unsigned short* W2m  = allocu((size_t)HDIM * upk(128));
  unsigned short* PW1t = allocu((size_t)POOLH1 * upk(92));      // [512][96]
  unsigned short* PW2t = allocu((size_t)POOLB * upk(POOLH1));   // [128][512]
  (void)ws_size; (void)out_size;

  // ---- pack weights (BT[n][k] = W[n][koff+k]) ----
  launch_pack(stream, p.ep_wih, wihP,  GATES, EDIM,  upk(EDIM),  EDIM,  0);
  launch_pack(stream, p.ep_whh, whhP,  GATES, HDIM,  upk(HDIM),  HDIM,  0);
  launch_pack(stream, p.em_wih, wihM,  GATES, EMDIM, upk(EMDIM), EMDIM, 0);
  launch_pack(stream, p.em_whh, whhM,  GATES, HDIM,  upk(HDIM),  HDIM,  0);
  launch_pack(stream, p.dp_wih, dwihP, GATES, EDIM,  upk(EDIM),  EDIM,  0);
  launch_pack(stream, p.dp_whh, dwhhP, GATES, HDIM,  upk(HDIM),  HDIM,  0);
  launch_pack(stream, p.dm_wih, dwihM, GATES, EMDIM, upk(EMDIM), EMDIM, 0);
  launch_pack(stream, p.dm_whh, dwhhM, GATES, HDIM,  upk(HDIM),  HDIM,  0);
  launch_pack(stream, p.h2p_w,  H2P,   EDIM,  HDIM,  upk(HDIM),  HDIM,  0);
  launch_pack(stream, p.h2m_w,  H2M,   EMDIM, HDIM,  upk(HDIM),  HDIM,  0);
  launch_pack(stream, p.l1_w,   L1a,   HDIM,  HDIM,  upk(HDIM),  2*HDIM, 0);
  launch_pack(stream, p.l1_w,   L1b,   HDIM,  HDIM,  upk(HDIM),  2*HDIM, HDIM);
  launch_pack(stream, p.m_w1,   W1a,   128,   HDIM,  upk(HDIM),  HDIM+POOLB, 0);
  launch_pack(stream, p.m_w1,   W1b,   128,   POOLB, upk(POOLB), HDIM+POOLB, HDIM);
  launch_pack(stream, p.m_w2,   W2m,   HDIM,  128,   upk(128),   128, 0);
  launch_pack(stream, p.pool_w1,PW1t,  POOLH1, 92,   upk(92),    92,  0);
  launch_pack(stream, p.pool_w2,PW2t,  POOLB, POOLH1, upk(POOLH1), POOLH1, 0);

  add_vec_kernel<<<1, 256, 0, stream>>>(p.ep_bih, p.ep_bhh, b_ep, GATES);
  add_vec_kernel<<<1, 256, 0, stream>>>(p.em_bih, p.em_bhh, b_em, GATES);
  add_vec_kernel<<<1, 256, 0, stream>>>(p.dp_bih, p.dp_bhh, b_dp, GATES);
  add_vec_kernel<<<1, 256, 0, stream>>>(p.dm_bih, p.dm_bhh, b_dm, GATES);

  const int nBH = BATCH * HDIM;
  zero_f32_kernel<<<(nBH + 255) / 256, 256, 0, stream>>>(h_p, nBH);
  zero_f32_kernel<<<(nBH + 255) / 256, 256, 0, stream>>>(c_p, nBH);
  zero_f32_kernel<<<(nBH + 255) / 256, 256, 0, stream>>>(hmA, nBH);
  zero_f32_kernel<<<(nBH + 255) / 256, 256, 0, stream>>>(c_m, nBH);

  // ---- encoder: 16 sequential LSTM steps (gates via WMMA GEMM) ----
  for (int t = 0; t < OBSLEN; ++t) {
    const float* xs = p.obs_speed + (size_t)t * BATCH * EDIM;
    launch_gemm(stream, xs,  EDIM, wihP, b_ep,    gates, GATES, BATCH, GATES, EDIM, 0);
    launch_gemm(stream, h_p, HDIM, whhP, nullptr, gates, GATES, BATCH, GATES, HDIM, 1);
    lstm_update_kernel<<<(nBH + 255) / 256, 256, 0, stream>>>(gates, h_p, c_p, nBH);
    const float* xm = p.obs_mask + (size_t)t * BATCH * EMDIM;
    launch_gemm(stream, xm,  EMDIM, wihM, b_em,    gates, GATES, BATCH, GATES, EMDIM, 0);
    launch_gemm(stream, hmA, HDIM,  whhM, nullptr, gates, GATES, BATCH, GATES, HDIM,  1);
    lstm_update_kernel<<<(nBH + 255) / 256, 256, 0, stream>>>(gates, hmA, c_m, nBH);
  }

  // ---- init decode state from last observation ----
  hipMemcpyAsync(cpose, p.obs_pose  + (size_t)15 * BATCH * EDIM,  (size_t)BATCH * EDIM  * 4,
                 hipMemcpyDeviceToDevice, stream);
  hipMemcpyAsync(lspd,  p.obs_speed + (size_t)15 * BATCH * EDIM,  (size_t)BATCH * EDIM  * 4,
                 hipMemcpyDeviceToDevice, stream);
  hipMemcpyAsync(lmask, p.obs_mask  + (size_t)15 * BATCH * EMDIM, (size_t)BATCH * EMDIM * 4,
                 hipMemcpyDeviceToDevice, stream);

  float* out_sp = (float*)d_out;                               // [14 x 4096 x 28]
  float* out_mk = out_sp + (size_t)PREDL * BATCH * EDIM;       // [14 x 4096 x 14]
  const size_t poolLds = (256 * 96 + 256 * 64 + 128 * 64) * sizeof(unsigned short);

  float* h_m = hmA; float* h_m_alt = hmB;

  for (int t = 0; t < PREDL; ++t) {
    // social pooling (dominant compute, fully fused WMMA kernel)
    pool_kernel<<<NGROUP, 512, poolLds, stream>>>(cpose, h_p, PW1t, p.pool_b1,
                                                  PW2t, p.pool_b2, poolh);
    // MLP: x = relu(bn(concat(h_p, pool) @ w1^T + b1))
    launch_gemm(stream, h_p,   HDIM,  W1a, p.m_b1,  xbuf, 128, BATCH, 128, HDIM,  0);
    launch_gemm(stream, poolh, POOLB, W1b, nullptr, xbuf, 128, BATCH, 128, POOLB, 1);
    bn_relu_kernel<<<(BATCH * 128 + 255) / 256, 256, 0, stream>>>(
        xbuf, xbuf, p.m_g1, p.m_be1, p.m_m1, p.m_v1, BATCH * 128, 128);
    // h_p = relu(bn(x @ w2^T + b2))
    launch_gemm(stream, xbuf, 128, W2m, p.m_b2, tmp64, HDIM, BATCH, HDIM, 128, 0);
    bn_relu_kernel<<<(nBH + 255) / 256, 256, 0, stream>>>(
        tmp64, h_p, p.m_g2, p.m_be2, p.m_m2, p.m_v2, nBH, HDIM);
    // h_m = concat(h_m, h_p) @ l1^T + b
    launch_gemm(stream, h_m, HDIM, L1a, p.l1_b,  h_m_alt, HDIM, BATCH, HDIM, HDIM, 0);
    launch_gemm(stream, h_p, HDIM, L1b, nullptr, h_m_alt, HDIM, BATCH, HDIM, HDIM, 1);
    { float* tsw = h_m; h_m = h_m_alt; h_m_alt = tsw; }
    // dec_p LSTM + speed head
    launch_gemm(stream, lspd, EDIM, dwihP, b_dp,    gates, GATES, BATCH, GATES, EDIM, 0);
    launch_gemm(stream, h_p,  HDIM, dwhhP, nullptr, gates, GATES, BATCH, GATES, HDIM, 1);
    lstm_update_kernel<<<(nBH + 255) / 256, 256, 0, stream>>>(gates, h_p, c_p, nBH);
    launch_gemm(stream, h_p, HDIM, H2P, p.h2p_b, sraw, EDIM, BATCH, EDIM, HDIM, 0);
    h2p_epi_kernel<<<(BATCH * EDIM + 255) / 256, 256, 0, stream>>>(
        sraw, lspd, cpose, out_sp + (size_t)t * BATCH * EDIM, BATCH * EDIM);
    // dec_m LSTM + mask head (+ distance threshold using this step's pose)
    launch_gemm(stream, lmask, EMDIM, dwihM, b_dm,    gates, GATES, BATCH, GATES, EMDIM, 0);
    launch_gemm(stream, h_m,   HDIM,  dwhhM, nullptr, gates, GATES, BATCH, GATES, HDIM,  1);
    lstm_update_kernel<<<(nBH + 255) / 256, 256, 0, stream>>>(gates, h_m, c_m, nBH);
    launch_gemm(stream, h_m, HDIM, H2M, p.h2m_b, mraw, EMDIM, BATCH, EMDIM, HDIM, 0);
    h2m_epi_kernel<<<(BATCH * EMDIM + 255) / 256, 256, 0, stream>>>(
        mraw, cpose, lmask, out_mk + (size_t)t * BATCH * EMDIM, BATCH * EMDIM);
  }
}
